// MambaBlock_37134287241900
// MI455X (gfx1250) — compile-verified, matrix operand order checked by reading, not run on hardware
//
#include <hip/hip_runtime.h>
#include <hip/hip_bf16.h>
#include <math.h>

typedef _Float16 f16;
typedef __attribute__((ext_vector_type(16))) _Float16 v16h;
typedef __attribute__((ext_vector_type(8)))  _Float16 v8h;
typedef __attribute__((ext_vector_type(8)))  float    v8f;

#define B_SZ   2
#define SEQL   2048
#define DMODEL 1024
#define DSTATE 16
#define DCONV  4
#define DINNER 2048
#define DTRANK 64
#define MROWS  (B_SZ * SEQL)          // 4096
#define NXZ    (2 * DINNER)           // 4096
#define NPROJ  (DTRANK + 2 * DSTATE)  // 96

// ---------------------------------------------------------------------------
// f32 -> f16 conversion (elementwise)
// ---------------------------------------------------------------------------
__global__ void cvt_f32_f16(const float* __restrict__ s, f16* __restrict__ d, int n) {
    int i = blockIdx.x * blockDim.x + threadIdx.x;
    if (i < n) d[i] = (f16)s[i];
}

// W is K x N row-major (f32); Bt is N x K row-major (f16)
__global__ void transpose_cvt(const float* __restrict__ W, f16* __restrict__ Bt,
                              int K, int N) {
    int i = blockIdx.x * blockDim.x + threadIdx.x;
    if (i < K * N) {
        int k = i / N, n = i % N;
        Bt[(size_t)n * K + k] = (f16)W[i];
    }
}

// dt_in = proj[:, 0:DTRANK] -> f16
__global__ void extract_dtin(const float* __restrict__ proj, f16* __restrict__ d) {
    int i = blockIdx.x * blockDim.x + threadIdx.x;
    if (i < MROWS * DTRANK) {
        int r = i / DTRANK, j = i % DTRANK;
        d[i] = (f16)proj[(size_t)r * NPROJ + j];
    }
}

// ---------------------------------------------------------------------------
// WMMA GEMM: C[M,N] (f32, row-major) = A[M,K] (f16, row-major) * Bt[N,K]^T
//
// Block = 256 threads = 8 waves, stacked along M.  Wave tile = (MT*16) rows x
// (NT*16) cols -> MT*NT WMMAs per 32-wide K-step.  The NT*16 x 32 B-tile is
// shared by all 8 waves: it is staged into LDS once per K-step using
// GLOBAL_LOAD_ASYNC_TO_LDS_B128 (ASYNCcnt), then each wave reads its
// fragments back with ds_load_b128.  LDS rows padded to 40 f16 (80 B) so the
// 16-lane fragment reads stride 20 banks mod 64 -> conflict free.
//
// f16 A/B fragment element i of lane (mn, hf):  K = 16*(i/8) + 8*hf + (i%8)
// f32 C element r of lane (mn, hf):             M = r + 8*hf, N = mn
// ---------------------------------------------------------------------------
template <int MT, int NT>
__global__ __launch_bounds__(256) void gemm_wmma_kernel(
    const f16* __restrict__ A, const f16* __restrict__ Bt,
    float* __restrict__ C, int M, int N, int K) {
    (void)M;
    constexpr int BSTRIDE = 40;  // f16 per LDS row (32 data + 8 pad)
    __shared__ f16 ldsB[NT * 16 * BSTRIDE];

    const int lane = threadIdx.x & 31;
    const int wave = threadIdx.x >> 5;
    const int hf   = lane >> 4;   // 16-lane half
    const int mn   = lane & 15;   // row (A) / col (B) index within tile

    const int tileM = blockIdx.y * (8 * MT * 16) + wave * (MT * 16);
    const int tileN = blockIdx.x * (16 * NT);

    const v8f vzero = {0.f, 0.f, 0.f, 0.f, 0.f, 0.f, 0.f, 0.f};
    v8f acc[MT][NT];
#pragma unroll
    for (int mi = 0; mi < MT; ++mi)
#pragma unroll
        for (int t = 0; t < NT; ++t) acc[mi][t] = vzero;

    // async B-tile copy assignment: NT*64 16-byte chunks (n-row, k-chunk)
    const int  chunk  = wave * (NT * 8) + lane;     // lanes [0, NT*8) take part
    const int  bn     = chunk >> 2;                 // 0 .. NT*16-1
    const int  bkc    = chunk & 3;                  // k-chunk of 8 f16
    const bool doCopy = (lane < NT * 8);
    const unsigned ldsOff =
        (unsigned)(size_t)(&ldsB[bn * BSTRIDE + bkc * 8]);
    const f16* gsrc = Bt + (size_t)(tileN + bn) * K + bkc * 8;

    const f16* arow[MT];
#pragma unroll
    for (int mi = 0; mi < MT; ++mi)
        arow[mi] = A + (size_t)(tileM + mi * 16 + mn) * K;

    for (int k0 = 0; k0 < K; k0 += 32) {
        __syncthreads();  // previous iteration's LDS reads complete
        if (doCopy) {
            unsigned long long ga = (unsigned long long)(size_t)(gsrc + k0);
            asm volatile("global_load_async_to_lds_b128 %0, %1, off"
                         :: "v"(ldsOff), "v"(ga) : "memory");
        }
        // A fragments straight from global (contiguous b128 pairs)
        v16h afrag[MT];
#pragma unroll
        for (int mi = 0; mi < MT; ++mi) {
            v8h lo = *(const v8h*)(arow[mi] + k0 + 8 * hf);
            v8h hi = *(const v8h*)(arow[mi] + k0 + 16 + 8 * hf);
#pragma unroll
            for (int i = 0; i < 8; ++i) { afrag[mi][i] = lo[i]; afrag[mi][i + 8] = hi[i]; }
        }
        asm volatile("s_wait_asynccnt 0x0" ::: "memory");
        __syncthreads();  // B tile visible to all waves

#pragma unroll
        for (int t = 0; t < NT; ++t) {
            v16h bfrag;
            const f16* brow = &ldsB[(t * 16 + mn) * BSTRIDE];
            v8h lo = *(const v8h*)(brow + 8 * hf);
            v8h hi = *(const v8h*)(brow + 16 + 8 * hf);
#pragma unroll
            for (int i = 0; i < 8; ++i) { bfrag[i] = lo[i]; bfrag[i + 8] = hi[i]; }
#pragma unroll
            for (int mi = 0; mi < MT; ++mi) {
                acc[mi][t] = __builtin_amdgcn_wmma_f32_16x16x32_f16(
                    false, afrag[mi], false, bfrag, (short)0, acc[mi][t], false, false);
            }
        }
    }

#pragma unroll
    for (int mi = 0; mi < MT; ++mi)
#pragma unroll
        for (int t = 0; t < NT; ++t)
#pragma unroll
            for (int r = 0; r < 8; ++r)
                C[(size_t)(tileM + mi * 16 + r + 8 * hf) * N + tileN + t * 16 + mn] =
                    acc[mi][t][r];
}

// ---------------------------------------------------------------------------
// Depthwise causal conv (K=4) + bias + SiLU.  xz is MROWS x NXZ (xb = cols
// [0,DINNER)).  Writes f32 (for scan) and f16 (for the x-proj GEMM).
// ---------------------------------------------------------------------------
__global__ void conv_silu(const float* __restrict__ xz, const float* __restrict__ cw,
                          const float* __restrict__ cb, float* __restrict__ xbc,
                          f16* __restrict__ xbc16) {
    int i = blockIdx.x * blockDim.x + threadIdx.x;
    if (i >= MROWS * DINNER) return;
    int d = i % DINNER;
    int row = i / DINNER;
    int l = row % SEQL;
    int b = row / SEQL;
    float acc = cb[d];
#pragma unroll
    for (int k = 0; k < DCONV; ++k) {
        int ls = l - (DCONV - 1) + k;
        if (ls >= 0)
            acc += xz[(size_t)(b * SEQL + ls) * NXZ + d] * cw[d * DCONV + k];
    }
    float v = acc / (1.f + __expf(-acc));  // silu
    xbc[i] = v;
    xbc16[i] = (f16)v;
}

// dt = softplus(dt_pre + b_dt[d]), in place
__global__ void softplus_bias(float* __restrict__ dt, const float* __restrict__ bdt) {
    int i = blockIdx.x * blockDim.x + threadIdx.x;
    if (i >= MROWS * DINNER) return;
    int d = i % DINNER;
    float v = dt[i] + bdt[d];
    dt[i] = (v > 20.f) ? v : log1pf(__expf(v));
}

// ---------------------------------------------------------------------------
// Selective scan.  One lane per (b, d, s): 16 lanes cooperate per channel,
// cross-lane sum over the 16 states via shuffle-xor (stays inside a 16-lane
// group of the wave32).  Epilogue fuses  + D*x  and  * silu(z), writes f16.
// ---------------------------------------------------------------------------
__global__ void scan_kernel(const float* __restrict__ xbc, const float* __restrict__ dt,
                            const float* __restrict__ proj, const float* __restrict__ xz,
                            const float* __restrict__ A_log, const float* __restrict__ Dp,
                            f16* __restrict__ y16) {
    int tid = blockIdx.x * blockDim.x + threadIdx.x;
    if (tid >= B_SZ * DINNER * DSTATE) return;
    int s = tid & (DSTATE - 1);
    int p = tid >> 4;            // (b, d) pair
    int d = p % DINNER;
    int b = p / DINNER;

    float As = -__expf(A_log[d * DSTATE + s]);
    float Dd = Dp[d];
    float h = 0.f;

    for (int l = 0; l < SEQL; ++l) {
        size_t row = (size_t)b * SEQL + l;
        float x_t  = xbc[row * DINNER + d];
        float dt_t = dt[row * DINNER + d];
        float Bs   = proj[row * NPROJ + DTRANK + s];
        float Cs   = proj[row * NPROJ + DTRANK + DSTATE + s];
        float dA   = __expf(dt_t * As);
        h = dA * h + dt_t * Bs * x_t;
        float y = h * Cs;
        y += __shfl_xor(y, 1);
        y += __shfl_xor(y, 2);
        y += __shfl_xor(y, 4);
        y += __shfl_xor(y, 8);
        if (s == 0) {
            y += Dd * x_t;
            float zv = xz[row * NXZ + DINNER + d];
            y *= zv / (1.f + __expf(-zv));  // * silu(z)
            y16[row * DINNER + d] = (f16)y;
        }
    }
}

// ---------------------------------------------------------------------------
extern "C" void kernel_launch(void* const* d_in, const int* in_sizes, int n_in,
                              void* d_out, int out_size, void* d_ws, size_t ws_size,
                              hipStream_t stream) {
    (void)in_sizes; (void)n_in; (void)out_size; (void)ws_size;
    const float* x       = (const float*)d_in[0];
    const float* W_in    = (const float*)d_in[1];
    const float* conv_w  = (const float*)d_in[2];
    const float* conv_b  = (const float*)d_in[3];
    const float* W_xproj = (const float*)d_in[4];
    const float* W_dt    = (const float*)d_in[5];
    const float* b_dt    = (const float*)d_in[6];
    const float* A_log   = (const float*)d_in[7];
    const float* Dvec    = (const float*)d_in[8];
    const float* W_out   = (const float*)d_in[9];
    float* out = (float*)d_out;

    char* ws = (char*)d_ws;
    size_t off = 0;
    auto alloc = [&](size_t bytes) -> char* {
        char* pp = ws + off;
        off += (bytes + 255) & ~(size_t)255;
        return pp;
    };
    f16*   x16    = (f16*)  alloc((size_t)MROWS * DMODEL * 2);
    f16*   win_t  = (f16*)  alloc((size_t)NXZ * DMODEL * 2);
    float* xz     = (float*)alloc((size_t)MROWS * NXZ * 4);
    float* xbc    = (float*)alloc((size_t)MROWS * DINNER * 4);
    f16*   xbc16  = (f16*)  alloc((size_t)MROWS * DINNER * 2);
    f16*   wxp_t  = (f16*)  alloc((size_t)NPROJ * DINNER * 2);
    float* proj   = (float*)alloc((size_t)MROWS * NPROJ * 4);
    f16*   dtin16 = (f16*)  alloc((size_t)MROWS * DTRANK * 2);
    f16*   wdt_t  = (f16*)  alloc((size_t)DINNER * DTRANK * 2);
    float* dtbuf  = (float*)alloc((size_t)MROWS * DINNER * 4);
    f16*   y16    = (f16*)  alloc((size_t)MROWS * DINNER * 2);
    f16*   wout_t = (f16*)  alloc((size_t)DMODEL * DINNER * 2);

    dim3 blk(256);

    // operand conversions (f32 -> f16, weights transposed to [N x K])
    cvt_f32_f16<<<(MROWS * DMODEL + 255) / 256, blk, 0, stream>>>(x, x16, MROWS * DMODEL);
    transpose_cvt<<<(DMODEL * NXZ + 255) / 256, blk, 0, stream>>>(W_in, win_t, DMODEL, NXZ);
    transpose_cvt<<<(DINNER * NPROJ + 255) / 256, blk, 0, stream>>>(W_xproj, wxp_t, DINNER, NPROJ);
    transpose_cvt<<<(DTRANK * DINNER + 255) / 256, blk, 0, stream>>>(W_dt, wdt_t, DTRANK, DINNER);
    transpose_cvt<<<(DINNER * DMODEL + 255) / 256, blk, 0, stream>>>(W_out, wout_t, DINNER, DMODEL);

    // GEMM1: xz = x @ W_in            (4096 x 4096, K=1024)
    gemm_wmma_kernel<2, 4><<<dim3(NXZ / 64, MROWS / 256), blk, 0, stream>>>(
        x16, win_t, xz, MROWS, NXZ, DMODEL);

    // causal depthwise conv + SiLU
    conv_silu<<<(MROWS * DINNER + 255) / 256, blk, 0, stream>>>(xz, conv_w, conv_b, xbc, xbc16);

    // GEMM2: proj = xb @ W_xproj      (4096 x 96, K=2048)
    gemm_wmma_kernel<2, 2><<<dim3(NPROJ / 32, MROWS / 256), blk, 0, stream>>>(
        xbc16, wxp_t, proj, MROWS, NPROJ, DINNER);

    // dt_in slice -> f16
    extract_dtin<<<(MROWS * DTRANK + 255) / 256, blk, 0, stream>>>(proj, dtin16);

    // GEMM3: dt_pre = dt_in @ W_dt    (4096 x 2048, K=64)
    gemm_wmma_kernel<2, 4><<<dim3(DINNER / 64, MROWS / 256), blk, 0, stream>>>(
        dtin16, wdt_t, dtbuf, MROWS, DINNER, DTRANK);

    // dt = softplus(dt_pre + b_dt)
    softplus_bias<<<(MROWS * DINNER + 255) / 256, blk, 0, stream>>>(dtbuf, b_dt);

    // selective scan (fused +D*x and *silu(z) epilogue, f16 output)
    scan_kernel<<<(B_SZ * DINNER * DSTATE + 255) / 256, blk, 0, stream>>>(
        xbc, dtbuf, proj, xz, A_log, Dvec, y16);

    // GEMM4: out = y @ W_out          (4096 x 1024, K=2048)
    gemm_wmma_kernel<2, 4><<<dim3(DMODEL / 64, MROWS / 256), blk, 0, stream>>>(
        y16, wout_t, out, MROWS, DMODEL, DINNER);
}
